// GAT_GraphConv_49916109914463
// MI455X (gfx1250) — matrix operand order repaired and checked by reading, not compiled, not run on hardware
//
#include <hip/hip_runtime.h>

#define NN      10000
#define NE      320000
#define ETOT    (NE + NN)          // edges + self loops
#define DD      128
#define HH      4
#define HCC     512                // H*C
#define WCOLS   1024               // [Wl | Wr]
#define NEG_SLOPE 0.2f
#define LN_EPS  1e-5f

typedef __attribute__((ext_vector_type(16))) __bf16 v16bf;
typedef __attribute__((ext_vector_type(8)))  float  v8f;

// ---------- helpers ----------
__device__ __forceinline__ unsigned short f2bf(float f) {
  unsigned u = __float_as_uint(f);
  u += 0x7FFFu + ((u >> 16) & 1u);          // round-to-nearest-even
  return (unsigned short)(u >> 16);
}
// monotonic float->uint encoding so atomicMax(uint) == float max
__device__ __forceinline__ unsigned encf(float f) {
  unsigned u = __float_as_uint(f);
  return (u & 0x80000000u) ? ~u : (u | 0x80000000u);
}
__device__ __forceinline__ float decf(unsigned u) {
  unsigned b = (u & 0x80000000u) ? (u & 0x7FFFFFFFu) : ~u;
  return __uint_as_float(b);
}
__device__ __forceinline__ float lrelu(float x) { return x > 0.f ? x : NEG_SLOPE * x; }

// ---------- conversions / packing ----------
__global__ void k_f32_to_bf16(const float* __restrict__ in, unsigned short* __restrict__ out, int n) {
  int i = blockIdx.x * blockDim.x + threadIdx.x;
  if (i < n) out[i] = f2bf(in[i]);
}

// wt[j][k] = W[k][j] (bf16), j in [0,1024): j<512 -> Wl, else Wr.  biascat[j] likewise.
__global__ void k_prep_w(const float* __restrict__ Wl, const float* __restrict__ bl,
                         const float* __restrict__ Wr, const float* __restrict__ br,
                         unsigned short* __restrict__ wt, float* __restrict__ biascat) {
  int i = blockIdx.x * blockDim.x + threadIdx.x;           // i = j*128 + k
  if (i >= WCOLS * DD) return;
  int j = i >> 7, k = i & 127;
  float w = (j < HCC) ? Wl[k * HCC + j] : Wr[k * HCC + (j - HCC)];
  wt[j * DD + k] = f2bf(w);
  if (k == 0) biascat[j] = (j < HCC) ? bl[j] : br[j - HCC];
}

// ---------- WMMA GEMM: out[N,1024] = x[N,128] @ Wcat[128,1024] + biascat ----------
// one wave -> one 16x16 tile; K=128 -> 4x v_wmma_f32_16x16x32_bf16
__global__ void k_gemm_bf16_wmma(const unsigned short* __restrict__ xbf,
                                 const unsigned short* __restrict__ wt,      // [1024][128]
                                 const float* __restrict__ biascat,
                                 float* __restrict__ out) {                  // [N][1024]
  const int wid = blockIdx.x * 8 + (threadIdx.x >> 5);
  if (wid >= (NN / 16) * (WCOLS / 16)) return;             // uniform per wave
  const int mt = wid >> 6;                                  // 64 N-tiles
  const int nt = wid & 63;
  const int lane = threadIdx.x & 31;
  const int half = lane >> 4;                               // K-half select
  const int l15  = lane & 15;

  const unsigned short* arow = xbf + (size_t)(mt * 16 + l15) * DD;   // A row m
  const unsigned short* brow = wt  + (size_t)(nt * 16 + l15) * DD;   // B col n

  v8f acc = {};
#pragma unroll
  for (int k0 = 0; k0 < DD; k0 += 32) {
    union { v16bf v; uint4 q[2]; } A, B;
    // 16-bit A 16x32 layout: elems 0..7 -> K = half*8 + i ; elems 8..15 -> K = 16 + half*8 + i
    A.q[0] = *(const uint4*)(arow + k0 + half * 8);
    A.q[1] = *(const uint4*)(arow + k0 + 16 + half * 8);
    // 16-bit B 32x16 layout: lane = col, elems 0..7 -> K = half*8 + i ; elems 8..15 -> K = 16 + half*8 + i
    B.q[0] = *(const uint4*)(brow + k0 + half * 8);
    B.q[1] = *(const uint4*)(brow + k0 + 16 + half * 8);
    acc = __builtin_amdgcn_wmma_f32_16x16x32_bf16(false, A.v, false, B.v,
                                                  (short)0, acc, false, false);
  }
  // D layout: VGPR r -> row = half*8 + r, col = l15
  const int row0 = mt * 16 + half * 8;
  const int col  = nt * 16 + l15;
  const float b = biascat[col];
  float* o = out + col;
#pragma unroll
  for (int r = 0; r < 8; ++r) o[(size_t)(row0 + r) * WCOLS] = acc[r] + b;
}

// ---------- edge pass 1: logits + segment max (wave per edge) ----------
__global__ void k_edge_logits(const int* __restrict__ srcA, const int* __restrict__ dstA,
                              const float* __restrict__ xlxr, const float* __restrict__ att,
                              float* __restrict__ ebuf, unsigned* __restrict__ menc) {
  int wid = blockIdx.x * 8 + (threadIdx.x >> 5);
  if (wid >= ETOT) return;
  int lane = threadIdx.x & 31;
  int s, d;
  if (wid < NE) { s = srcA[wid]; d = dstA[wid]; } else { s = d = wid - NE; }
  const float* xl = xlxr + (size_t)s * WCOLS;
  const float* xr = xlxr + (size_t)d * WCOLS + HCC;
  int c0 = lane * 4;
#pragma unroll
  for (int h = 0; h < HH; ++h) {
    float4 a = *(const float4*)(att + h * DD + c0);
    float4 l = *(const float4*)(xl + h * DD + c0);
    float4 r = *(const float4*)(xr + h * DD + c0);
    float p = a.x * lrelu(l.x + r.x) + a.y * lrelu(l.y + r.y)
            + a.z * lrelu(l.z + r.z) + a.w * lrelu(l.w + r.w);
#pragma unroll
    for (int m = 16; m >= 1; m >>= 1) p += __shfl_xor(p, m, 32);
    if (lane == 0) {
      ebuf[(size_t)wid * HH + h] = p;
      atomicMax(&menc[d * HH + h], encf(p));
    }
  }
}

// ---------- edge pass 2: exp(e - m), denom sum (thread per edge*head) ----------
__global__ void k_edge_exp(const int* __restrict__ dstA, const unsigned* __restrict__ menc,
                           float* __restrict__ ebuf, float* __restrict__ denom) {
  int idx = blockIdx.x * blockDim.x + threadIdx.x;
  if (idx >= ETOT * HH) return;
  int e = idx >> 2, h = idx & 3;
  int d = (e < NE) ? dstA[e] : (e - NE);
  float m = decf(menc[d * HH + h]);
  float ex = __expf(ebuf[idx] - m);
  ebuf[idx] = ex;
  atomicAdd(&denom[d * HH + h], ex);
}

// ---------- edge pass 3: out[dst] += alpha * xl[src] (wave per edge) ----------
__global__ void k_edge_aggr(const int* __restrict__ srcA, const int* __restrict__ dstA,
                            const float* __restrict__ xlxr, const float* __restrict__ ebuf,
                            const float* __restrict__ denom, float* __restrict__ outacc) {
  int wid = blockIdx.x * 8 + (threadIdx.x >> 5);
  if (wid >= ETOT) return;
  int lane = threadIdx.x & 31;
  int s, d;
  if (wid < NE) { s = srcA[wid]; d = dstA[wid]; } else { s = d = wid - NE; }
  const float* xl = xlxr + (size_t)s * WCOLS;
  float* oa = outacc + (size_t)d * HCC;
  int c0 = lane * 4;
#pragma unroll
  for (int h = 0; h < HH; ++h) {
    float alpha = ebuf[(size_t)wid * HH + h] / denom[d * HH + h];
    float4 l = *(const float4*)(xl + h * DD + c0);
    atomicAdd(oa + h * DD + c0 + 0, alpha * l.x);
    atomicAdd(oa + h * DD + c0 + 1, alpha * l.y);
    atomicAdd(oa + h * DD + c0 + 2, alpha * l.z);
    atomicAdd(oa + h * DD + c0 + 3, alpha * l.w);
  }
}

// ---------- finalize: head-mean + bias + residual, LayerNorm, ReLU ----------
__global__ void k_finalize(const float* __restrict__ outacc, const float* __restrict__ xold,
                           const float* __restrict__ bias, const float* __restrict__ g,
                           const float* __restrict__ b, float* __restrict__ xnew) {
  int node = blockIdx.x, c = threadIdx.x;   // 128 threads
  const float* oa = outacc + (size_t)node * HCC;
  float v = 0.25f * (oa[c] + oa[DD + c] + oa[2 * DD + c] + oa[3 * DD + c])
          + bias[c] + xold[(size_t)node * DD + c];
  __shared__ float red[DD];
  red[c] = v; __syncthreads();
#pragma unroll
  for (int s = 64; s > 0; s >>= 1) { if (c < s) red[c] += red[c + s]; __syncthreads(); }
  float mu = red[0] * (1.f / DD);
  __syncthreads();
  float dv = v - mu;
  red[c] = dv * dv; __syncthreads();
#pragma unroll
  for (int s = 64; s > 0; s >>= 1) { if (c < s) red[c] += red[c + s]; __syncthreads(); }
  float var = red[0] * (1.f / DD);
  float o = dv * rsqrtf(var + LN_EPS) * g[c] + b[c];
  xnew[(size_t)node * DD + c] = fmaxf(o, 0.f);
}

// ---------- host ----------
extern "C" void kernel_launch(void* const* d_in, const int* in_sizes, int n_in,
                              void* d_out, int out_size, void* d_ws, size_t ws_size,
                              hipStream_t stream) {
  const float* x0   = (const float*)d_in[0];
  const int*   srcA = (const int*)d_in[1];
  const int*   dstA = srcA + NE;
  const float* Wl   = (const float*)d_in[2];
  const float* bl   = (const float*)d_in[3];
  const float* Wr   = (const float*)d_in[4];
  const float* br   = (const float*)d_in[5];
  const float* att  = (const float*)d_in[6];
  const float* bias = (const float*)d_in[7];
  const float* lng  = (const float*)d_in[8];
  const float* lnb  = (const float*)d_in[9];

  char* p = (char*)d_ws;
  auto alloc = [&](size_t bytes) -> char* {
    char* r = p; p += (bytes + 255) & ~(size_t)255; return r;
  };
  float*          x_cur   = (float*)alloc((size_t)NN * DD * 4);
  unsigned short* xbf     = (unsigned short*)alloc((size_t)NN * DD * 2);
  unsigned short* wt      = (unsigned short*)alloc((size_t)WCOLS * DD * 2);
  float*          biascat = (float*)alloc(WCOLS * 4);
  float*          xlxr    = (float*)alloc((size_t)NN * WCOLS * 4);
  float*          ebuf    = (float*)alloc((size_t)ETOT * HH * 4);
  char* zbase = p;                                   // contiguous zero-init region
  unsigned*       menc    = (unsigned*)alloc((size_t)NN * HH * 4);
  float*          denom   = (float*)alloc((size_t)NN * HH * 4);
  float*          outacc  = (float*)alloc((size_t)NN * HCC * 4);
  size_t zbytes = (size_t)(p - zbase);

  for (int l = 0; l < 2; ++l) {
    const float* xin  = (l == 0) ? x0 : x_cur;
    float*       xout = (l == 1) ? (float*)d_out : x_cur;

    k_f32_to_bf16<<<(NN * DD + 255) / 256, 256, 0, stream>>>(xin, xbf, NN * DD);
    k_prep_w<<<(WCOLS * DD + 255) / 256, 256, 0, stream>>>(
        Wl + (size_t)l * DD * HCC, bl + (size_t)l * HCC,
        Wr + (size_t)l * DD * HCC, br + (size_t)l * HCC, wt, biascat);
    k_gemm_bf16_wmma<<<(NN / 16) * (WCOLS / 16) / 8, 256, 0, stream>>>(xbf, wt, biascat, xlxr);
    hipMemsetAsync(zbase, 0, zbytes, stream);
    k_edge_logits<<<(ETOT + 7) / 8, 256, 0, stream>>>(srcA, dstA, xlxr,
                                                      att + (size_t)l * HH * DD, ebuf, menc);
    k_edge_exp<<<(ETOT * HH + 255) / 256, 256, 0, stream>>>(dstA, menc, ebuf, denom);
    k_edge_aggr<<<(ETOT + 7) / 8, 256, 0, stream>>>(srcA, dstA, xlxr, ebuf, denom, outacc);
    k_finalize<<<NN, DD, 0, stream>>>(outacc, xin, bias + (size_t)l * DD,
                                      lng + (size_t)l * DD, lnb + (size_t)l * DD, xout);
  }
}
